// S5Dual_30975304139517
// MI455X (gfx1250) — matrix operand order repaired and checked by reading, not compiled, not run on hardware
//
#include <hip/hip_runtime.h>
#include <hip/hip_fp16.h>
#include <math.h>

typedef __attribute__((ext_vector_type(16))) _Float16 v16h;
typedef __attribute__((ext_vector_type(8)))  _Float16 v8h;
typedef __attribute__((ext_vector_type(8)))  float    v8f;
typedef __attribute__((__vector_size__(16))) int      v4i_;

namespace {
constexpr int SEQ = 8192;
constexpr int HD  = 512;
constexpr int PD  = 512;
constexpr int NCPLX = 2 * PD;        // 1024 complex channels (2 dirs x P)
constexpr int BUC = 4 * PD;          // 2048 real columns of Bu/xs
constexpr int CHUNK = 128;
constexpr int NCHUNK = SEQ / CHUNK;  // 64
}

// ---------------------------------------------------------------------------
// CDNA5 async global->LDS copy (ASYNCcnt-tracked), 16 bytes per lane.
// ---------------------------------------------------------------------------
__device__ __forceinline__ void async_copy16(const _Float16* g, _Float16* l) {
#if __has_builtin(__builtin_amdgcn_global_load_async_to_lds_b128)
  __builtin_amdgcn_global_load_async_to_lds_b128(
      (__attribute__((address_space(1))) v4i_*)(void*)(const void*)g,
      (__attribute__((address_space(3))) v4i_*)(void*)l, 0, 0);
#else
  // Generic pointer to LDS: low 32 bits are the LDS byte offset (aperture rule).
  unsigned loff = (unsigned)(unsigned long long)l;
  asm volatile("global_load_async_to_lds_b128 %0, %1, off offset:0"
               :: "v"(loff), "v"(g) : "memory");
#endif
}

__device__ __forceinline__ void wait_async_le4() {
#if __has_builtin(__builtin_amdgcn_s_wait_asynccnt)
  __builtin_amdgcn_s_wait_asynccnt(4);
#else
  asm volatile("s_wait_asynccnt 0x4" ::: "memory");
#endif
}

__device__ __forceinline__ void wait_async_le0() {
#if __has_builtin(__builtin_amdgcn_s_wait_asynccnt)
  __builtin_amdgcn_s_wait_asynccnt(0);
#else
  asm volatile("s_wait_asynccnt 0x0" ::: "memory");
#endif
}

// ---------------------------------------------------------------------------
// Param prep: Lambda_bar and B_bar scale per (dir, p)
// ---------------------------------------------------------------------------
__global__ void k_prep_lambda(const float* __restrict__ log_real,
                              const float* __restrict__ imag,
                              const float* __restrict__ log_Delta,
                              float* __restrict__ lam,      // (2*P) complex
                              float* __restrict__ scale) {  // (2*P) complex
  int i = blockIdx.x * blockDim.x + threadIdx.x;
  if (i >= NCPLX) return;
  float lr = -__expf(log_real[i]);
  float li = imag[i];
  float dt = __expf(log_Delta[i]);
  float er = __expf(lr * dt);
  float lbr = er * __cosf(li * dt);
  float lbi = er * __sinf(li * dt);
  lam[2 * i]     = lbr;
  lam[2 * i + 1] = lbi;
  float mag = sqrtf(lr * lr + li * li);
  float sr = lr, si = li;
  if (mag < 1e-6f) { sr = 1e-6f; si = 0.0f; }
  float den = sr * sr + si * si;
  float nr = lbr - 1.0f, ni = lbi;
  scale[2 * i]     = (nr * sr + ni * si) / den;   // (lb-1)/Lambda_safe
  scale[2 * i + 1] = (ni * sr - nr * si) / den;
}

// W1[(dir*1024 + comp*512 + p)][h] = f16(B_bar component), row-major K=H
__global__ void k_prep_w1(const float* __restrict__ Breal,
                          const float* __restrict__ Bimag,
                          const float* __restrict__ scale,
                          _Float16* __restrict__ W1) {
  int idx = blockIdx.x * 256 + threadIdx.x;     // over 2*P*H
  if (idx >= 2 * PD * HD) return;
  int h   = idx & (HD - 1);
  int p   = (idx >> 9) & (PD - 1);
  int dir = idx >> 18;
  int pp  = dir * PD + p;
  float br = Breal[(size_t)pp * HD + h];
  float bi = Bimag[(size_t)pp * HD + h];
  float sr = scale[2 * pp], si = scale[2 * pp + 1];
  W1[((size_t)dir * 1024 + p) * HD + h]       = (_Float16)(sr * br - si * bi);
  W1[((size_t)dir * 1024 + 512 + p) * HD + h] = (_Float16)(sr * bi + si * br);
}

// W2[dir][h][p] = C_r ; W2[dir][h][512+p] = -C_i  (row-major K=1024)
__global__ void k_prep_w2(const float* __restrict__ Creal,
                          const float* __restrict__ Cimag,
                          _Float16* __restrict__ W2) {
  int idx = blockIdx.x * 256 + threadIdx.x;     // over 2*H*P
  if (idx >= 2 * HD * PD) return;
  int p   = idx & (PD - 1);
  int h   = (idx >> 9) & (HD - 1);
  int dir = idx >> 18;
  size_t row = (size_t)dir * HD + h;
  W2[row * 1024 + p]       = (_Float16)( Creal[row * PD + p]);
  W2[row * 1024 + 512 + p] = (_Float16)(-Cimag[row * PD + p]);
}

__global__ void k_prep_xh(const float* __restrict__ x, _Float16* __restrict__ xh) {
  int idx = blockIdx.x * 256 + threadIdx.x;
  if (idx < SEQ * HD) xh[idx] = (_Float16)x[idx];
}

// ---------------------------------------------------------------------------
// Tiled f16 WMMA GEMM: Cout(M x N) = A(M x K) * W(N x K)^T , fp32 accumulate.
// 256 threads = 8 wave32; 128x128 tile; K-step 32; LDS row stride 40 halves.
// Double-buffered async global->LDS pipeline (ASYNCcnt).
// ---------------------------------------------------------------------------
#define LDT 40

__global__ __launch_bounds__(256)
void k_gemm_f16(const _Float16* __restrict__ A, int lda,
                const _Float16* __restrict__ W, int ldw,
                float* __restrict__ Cout, int ldc, int K) {
  __shared__ _Float16 sA[2][128 * LDT];
  __shared__ _Float16 sB[2][128 * LDT];

  const int tid  = threadIdx.x;
  const int lane = tid & 31;
  const int wave = tid >> 5;     // 0..7
  const int wm   = wave >> 1;    // 0..3  (M sub-tile)
  const int wn   = wave & 1;     // 0..1  (N sub-tile)
  const int m0   = blockIdx.y * 128;
  const int n0   = blockIdx.x * 128;

  const int lr_ = tid >> 1;            // 0..127: tile row this thread loads
  const int lc_ = (tid & 1) * 16;      // 0 or 16 halves

  const int mrow = lane & 15;
  const int hi   = lane >> 4;

  v8f acc[2][4] = {};

  // Issue one K-tile (4 x b128 async per thread: 2 for A, 2 for B).
  auto issue = [&](int buf, int k0) {
    const _Float16* gA = A + (size_t)(m0 + lr_) * lda + k0 + lc_;
    const _Float16* gB = W + (size_t)(n0 + lr_) * ldw + k0 + lc_;
    _Float16* la = &sA[buf][lr_ * LDT + lc_];
    _Float16* lb = &sB[buf][lr_ * LDT + lc_];
    async_copy16(gA,     la);
    async_copy16(gA + 8, la + 8);
    async_copy16(gB,     lb);
    async_copy16(gB + 8, lb + 8);
  };

  const int nk = K / 32;
  issue(0, 0);

  for (int ik = 0; ik < nk; ++ik) {
    if (ik + 1 < nk) {
      issue((ik + 1) & 1, (ik + 1) * 32);   // prefetch next tile into other buf
      wait_async_le4();                     // current tile's 4 asyncs complete
    } else {
      wait_async_le0();
    }
    __syncthreads();                        // all threads' asyncs visible

    const _Float16* bufA = sA[ik & 1];
    const _Float16* bufB = sB[ik & 1];

    v16h af[2], bf[4];
#pragma unroll
    for (int i = 0; i < 2; ++i) {
      // A 16x32 frag: lanes 0-15 hold K 0-7 & 16-23; lanes 16-31 hold 8-15 & 24-31
      const _Float16* pa = bufA + (wm * 32 + i * 16 + mrow) * LDT + hi * 8;
      v8h lo = *(const v8h*)pa;
      v8h hi8 = *(const v8h*)(pa + 16);
      af[i] = __builtin_shufflevector(lo, hi8, 0, 1, 2, 3, 4, 5, 6, 7,
                                      8, 9, 10, 11, 12, 13, 14, 15);
    }
#pragma unroll
    for (int j = 0; j < 4; ++j) {
      // B 32x16 frag: lanes 0-15 hold K 0-7 & 16-23; lanes 16-31 hold 8-15 & 24-31
      const _Float16* pb = bufB + (wn * 64 + j * 16 + mrow) * LDT + hi * 8;
      v8h lo = *(const v8h*)pb;
      v8h hi8 = *(const v8h*)(pb + 16);
      bf[j] = __builtin_shufflevector(lo, hi8, 0, 1, 2, 3, 4, 5, 6, 7,
                                      8, 9, 10, 11, 12, 13, 14, 15);
    }
#pragma unroll
    for (int i = 0; i < 2; ++i)
#pragma unroll
      for (int j = 0; j < 4; ++j)
        acc[i][j] = __builtin_amdgcn_wmma_f32_16x16x32_f16(
            false, af[i], false, bf[j], (short)0, acc[i][j], false, false);

    __syncthreads();   // done reading this buf before it is refilled (ik+2)
  }

#pragma unroll
  for (int i = 0; i < 2; ++i)
#pragma unroll
    for (int j = 0; j < 4; ++j) {
      const int gn = n0 + wn * 64 + j * 16 + mrow;
#pragma unroll
      for (int e = 0; e < 8; ++e) {
        const int gm = m0 + wm * 32 + i * 16 + hi * 8 + e;
        Cout[(size_t)gm * ldc + gn] = acc[i][j][e];
      }
    }
}

// ---------------------------------------------------------------------------
// 3-phase chunked complex diagonal scan over L (dir0 forward, dir1 reverse)
// ---------------------------------------------------------------------------
__global__ void k_scan1(float* __restrict__ Bu, const float* __restrict__ lam,
                        float* __restrict__ agg) {
  int ch    = blockIdx.y * blockDim.x + threadIdx.x;  // 0..1023 complex channel
  int chunk = blockIdx.x;                             // 0..63
  int dir = ch >> 9, p = ch & (PD - 1);
  float lr = lam[2 * ch], li = lam[2 * ch + 1];
  int cr = dir * 1024 + p, ci = cr + 512;
  float xr = 0.f, xi = 0.f;
  int tv0 = chunk * CHUNK;
  for (int t = 0; t < CHUNK; ++t) {
    int tv  = tv0 + t;
    int row = dir ? (SEQ - 1 - tv) : tv;
    float* pr = Bu + (size_t)row * BUC;
    float br = pr[cr], bi = pr[ci];
    float nxr = lr * xr - li * xi + br;
    float nxi = lr * xi + li * xr + bi;
    xr = nxr; xi = nxi;
    pr[cr] = xr; pr[ci] = xi;      // local inclusive scan in place
  }
  agg[(size_t)(chunk * NCPLX + ch) * 2]     = xr;
  agg[(size_t)(chunk * NCPLX + ch) * 2 + 1] = xi;
}

__global__ void k_scan2(const float* __restrict__ lam,
                        const float* __restrict__ agg,
                        float* __restrict__ carry) {
  int ch = blockIdx.x * blockDim.x + threadIdx.x;
  if (ch >= NCPLX) return;
  float lr = lam[2 * ch], li = lam[2 * ch + 1];
  float fr = lr, fi = li;
#pragma unroll
  for (int s = 0; s < 7; ++s) {  // Lambda^128
    float nr = fr * fr - fi * fi;
    float ni = 2.f * fr * fi;
    fr = nr; fi = ni;
  }
  float cr_ = 0.f, ci_ = 0.f;
  for (int c = 0; c < NCHUNK; ++c) {
    carry[(size_t)(c * NCPLX + ch) * 2]     = cr_;   // exclusive
    carry[(size_t)(c * NCPLX + ch) * 2 + 1] = ci_;
    float ar = agg[(size_t)(c * NCPLX + ch) * 2];
    float ai = agg[(size_t)(c * NCPLX + ch) * 2 + 1];
    float nr = fr * cr_ - fi * ci_ + ar;
    float ni = fr * ci_ + fi * cr_ + ai;
    cr_ = nr; ci_ = ni;
  }
}

__global__ void k_scan3(const float* __restrict__ Bu,
                        const float* __restrict__ lam,
                        const float* __restrict__ carry,
                        _Float16* __restrict__ xsh) {
  int ch    = blockIdx.y * blockDim.x + threadIdx.x;
  int chunk = blockIdx.x;
  int dir = ch >> 9, p = ch & (PD - 1);
  float lr = lam[2 * ch], li = lam[2 * ch + 1];
  int cr = dir * 1024 + p, ci = cr + 512;
  float kr = carry[(size_t)(chunk * NCPLX + ch) * 2];
  float ki = carry[(size_t)(chunk * NCPLX + ch) * 2 + 1];
  float wr = lr * kr - li * ki;      // Lambda^1 * carry
  float wi = lr * ki + li * kr;
  int tv0 = chunk * CHUNK;
  for (int t = 0; t < CHUNK; ++t) {
    int tv  = tv0 + t;
    int row = dir ? (SEQ - 1 - tv) : tv;
    const float* pr = Bu + (size_t)row * BUC;
    float xr = pr[cr] + wr;
    float xi = pr[ci] + wi;
    _Float16* ph = xsh + (size_t)row * BUC;
    ph[cr] = (_Float16)xr;
    ph[ci] = (_Float16)xi;
    float nwr = lr * wr - li * wi;   // w *= Lambda
    float nwi = lr * wi + li * wr;
    wr = nwr; wi = nwi;
  }
}

// ---------------------------------------------------------------------------
// Epilogue: out = gelu(ys_f + x*D0) + gelu(ys_b + x*D1)   (tanh gelu, per jax)
// ---------------------------------------------------------------------------
__device__ __forceinline__ float gelu_tanh(float v) {
  float c = 0.7978845608028654f * (v + 0.044715f * v * v * v);
  return 0.5f * v * (1.0f + tanhf(c));
}

__global__ void k_final(const float* __restrict__ ys, const float* __restrict__ x,
                        const float* __restrict__ D, float* __restrict__ out) {
  int idx = blockIdx.x * 256 + threadIdx.x;
  if (idx >= SEQ * HD) return;
  int h = idx & (HD - 1);
  int t = idx >> 9;
  float xv = x[idx];
  float a = ys[(size_t)t * 1024 + h]       + xv * D[h];
  float b = ys[(size_t)t * 1024 + 512 + h] + xv * D[HD + h];
  out[idx] = gelu_tanh(a) + gelu_tanh(b);
}

// ---------------------------------------------------------------------------
extern "C" void kernel_launch(void* const* d_in, const int* in_sizes, int n_in,
                              void* d_out, int out_size, void* d_ws, size_t ws_size,
                              hipStream_t stream) {
  (void)in_sizes; (void)n_in; (void)out_size; (void)ws_size;
  const float* x        = (const float*)d_in[0];
  const float* log_real = (const float*)d_in[1];
  const float* imag     = (const float*)d_in[2];
  const float* B_real   = (const float*)d_in[3];
  const float* B_imag   = (const float*)d_in[4];
  const float* C_real   = (const float*)d_in[5];
  const float* C_imag   = (const float*)d_in[6];
  const float* Dm       = (const float*)d_in[7];
  const float* log_Dt   = (const float*)d_in[8];
  float* out = (float*)d_out;

  char* ws = (char*)d_ws;
  size_t off = 0;
  _Float16* xh  = (_Float16*)(ws + off); off += (size_t)SEQ * HD * 2;        // 8 MB
  _Float16* W1  = (_Float16*)(ws + off); off += (size_t)BUC * HD * 2;        // 2 MB
  _Float16* W2  = (_Float16*)(ws + off); off += (size_t)2 * HD * 1024 * 2;   // 2 MB
  float* lam    = (float*)(ws + off);    off += (size_t)NCPLX * 2 * 4;
  float* scale  = (float*)(ws + off);    off += (size_t)NCPLX * 2 * 4;
  float* agg    = (float*)(ws + off);    off += (size_t)NCHUNK * NCPLX * 2 * 4;
  float* carry  = (float*)(ws + off);    off += (size_t)NCHUNK * NCPLX * 2 * 4;
  float* Bu     = (float*)(ws + off);    off += (size_t)SEQ * BUC * 4;       // 64 MB
  _Float16* xsh = (_Float16*)(ws + off); off += (size_t)SEQ * BUC * 2;       // 32 MB
  float* ys = Bu;  // reuse Bu region after scan consumes it (33.5 MB <= 64 MB)

  // 1) parameter prep
  k_prep_lambda<<<(NCPLX + 255) / 256, 256, 0, stream>>>(log_real, imag, log_Dt,
                                                         lam, scale);
  k_prep_w1<<<(2 * PD * HD) / 256, 256, 0, stream>>>(B_real, B_imag, scale, W1);
  k_prep_w2<<<(2 * HD * PD) / 256, 256, 0, stream>>>(C_real, C_imag, W2);
  k_prep_xh<<<(SEQ * HD) / 256, 256, 0, stream>>>(x, xh);

  // 2) input projection: Bu(8192 x 2048) = xh(8192x512) @ W1(2048x512)^T
  k_gemm_f16<<<dim3(BUC / 128, SEQ / 128), 256, 0, stream>>>(
      xh, HD, W1, HD, Bu, BUC, HD);

  // 3) diagonal complex scan (dir0 fwd, dir1 reversed-time), 3 phases
  k_scan1<<<dim3(NCHUNK, NCPLX / 128), 128, 0, stream>>>(Bu, lam, agg);
  k_scan2<<<(NCPLX + 255) / 256, 256, 0, stream>>>(lam, agg, carry);
  k_scan3<<<dim3(NCHUNK, NCPLX / 128), 128, 0, stream>>>(Bu, lam, carry, xsh);

  // 4) output projection per direction: ys(:, dir*512 + h)
  k_gemm_f16<<<dim3(HD / 128, SEQ / 128), 256, 0, stream>>>(
      xsh, BUC, W2, 1024, ys, 1024, 1024);
  k_gemm_f16<<<dim3(HD / 128, SEQ / 128), 256, 0, stream>>>(
      xsh + 1024, BUC, W2 + (size_t)HD * 1024, 1024, ys + 512, 1024, 1024);

  // 5) feedthrough + gelu + combine directions
  k_final<<<(SEQ * HD) / 256, 256, 0, stream>>>(ys, x, Dm, out);
}
